// AltBlock_54829552501377
// MI455X (gfx1250) — matrix operand order repaired and checked by reading, not compiled, not run on hardware
//
#include <hip/hip_runtime.h>

// ---------------------------------------------------------------------------
// CDNA5 (gfx1250, wave32) transformer block: LN -> QKV -> ALiBi attention ->
// proj(+scale/bias+res) -> LN -> GLU-MLP -> (+scale/bias+res).
// GEMMs via v_wmma_f32_16x16x32_bf16; attention K/V staged into LDS with
// global_load_async_to_lds_b128 (ASYNCcnt) and consumed as WMMA fragments.
// ---------------------------------------------------------------------------

typedef __attribute__((ext_vector_type(16))) __bf16 v16bf;
typedef __attribute__((ext_vector_type(8)))  __bf16 v8bf;
typedef __attribute__((ext_vector_type(8)))  float  v8f;
typedef __attribute__((ext_vector_type(4)))  float  v4f;

static constexpr int BATCH = 8;
static constexpr int S     = 2048;
static constexpr int D     = 256;
static constexpr int H     = 4;
static constexpr int HD    = 64;       // head dim
static constexpr int DFF   = 1024;
static constexpr int DH    = 512;      // DFF / 2
static constexpr int TOK   = BATCH * S;   // 16384 rows
static constexpr int QKVW  = 3 * D;       // 768

#define DEV __device__ __forceinline__

DEV unsigned short f2bf(float f) {
  unsigned int u = __float_as_uint(f);
  u += 0x7FFFu + ((u >> 16) & 1u);          // round-to-nearest-even
  return (unsigned short)(u >> 16);
}

// One async 16B-per-lane copy global -> LDS (wave instruction, ASYNCcnt).
DEV void async_b128(unsigned lds_off, const void* gptr) {
  unsigned long long ga = (unsigned long long)(uintptr_t)gptr;
  asm volatile("global_load_async_to_lds_b128 %0, %1, off"
               :: "v"(lds_off), "v"(ga) : "memory");
}

// ---------------------------------------------------------------------------
// LayerNorm (f32 in) -> bf16 out.  One wave32 per 256-wide row.
// ---------------------------------------------------------------------------
__global__ __launch_bounds__(128) void ln_bf16_kernel(
    const float* __restrict__ x, const float* __restrict__ gamma,
    const float* __restrict__ beta, unsigned short* __restrict__ out) {
  const int lane = threadIdx.x & 31;
  const int row  = blockIdx.x * 4 + (threadIdx.x >> 5);
  const float* xr = x + (size_t)row * D + lane * 8;
  v4f a = *(const v4f*)xr;
  v4f b = *(const v4f*)(xr + 4);
  float s = a[0] + a[1] + a[2] + a[3] + b[0] + b[1] + b[2] + b[3];
#pragma unroll
  for (int d = 1; d < 32; d <<= 1) s += __shfl_xor(s, d, 32);
  const float mu = s * (1.0f / D);
  float vs = 0.f;
#pragma unroll
  for (int i = 0; i < 4; ++i) { float t = a[i] - mu; vs += t * t; }
#pragma unroll
  for (int i = 0; i < 4; ++i) { float t = b[i] - mu; vs += t * t; }
#pragma unroll
  for (int d = 1; d < 32; d <<= 1) vs += __shfl_xor(vs, d, 32);
  const float inv = rsqrtf(vs * (1.0f / D) + 1e-5f);
  unsigned short* orow = out + (size_t)row * D + lane * 8;
#pragma unroll
  for (int i = 0; i < 4; ++i)
    orow[i] = f2bf((a[i] - mu) * inv * gamma[lane * 8 + i] + beta[lane * 8 + i]);
#pragma unroll
  for (int i = 0; i < 4; ++i)
    orow[4 + i] = f2bf((b[i] - mu) * inv * gamma[lane * 8 + 4 + i] + beta[lane * 8 + 4 + i]);
}

// ---------------------------------------------------------------------------
// Weight convert: f32 [K,N] row-major -> bf16 W^T [N,K]
// ---------------------------------------------------------------------------
__global__ void wconv_kernel(const float* __restrict__ w,
                             unsigned short* __restrict__ wt, int K, int N) {
  int idx = blockIdx.x * blockDim.x + threadIdx.x;
  if (idx >= K * N) return;
  int k = idx / N, n = idx % N;
  wt[(size_t)n * K + k] = f2bf(w[idx]);
}

// V transpose: qkv bf16 -> vT[b][h][d][s]
__global__ void vtrans_kernel(const unsigned short* __restrict__ qkvb,
                              unsigned short* __restrict__ vT) {
  int idx = blockIdx.x * blockDim.x + threadIdx.x;   // 2^22 total
  int s = idx & (S - 1);
  int d = (idx >> 11) & (HD - 1);
  int h = (idx >> 17) & (H - 1);
  int b = idx >> 19;
  vT[idx] = qkvb[((size_t)(b * S + s)) * QKVW + h * 192 + 128 + d];
}

// GLU: h f32 [TOK,1024] -> out * gelu(gate) -> bf16 [TOK,512]
__global__ void glu_kernel(const float* __restrict__ hbuf,
                           unsigned short* __restrict__ gbuf) {
  int idx = blockIdx.x * blockDim.x + threadIdx.x;
  int row = idx >> 9;
  int c   = idx & 511;
  float o  = hbuf[(size_t)row * DFF + c];
  float gt = hbuf[(size_t)row * DFF + DH + c];
  float ge = 0.5f * gt * (1.0f + erff(gt * 0.70710678118654752f));
  gbuf[idx] = f2bf(o * ge);
}

// ---------------------------------------------------------------------------
// GEMM: C[M,N] = A[M,K](bf16) * B(bf16, stored as B^T [N,K]) + bias
// One wave per 16(M) x 64(N) tile; 4 WMMAs per 32-wide K chunk.
// FUSED: C = C*scale[col] + sbias[col] + resid[row,col]
// ---------------------------------------------------------------------------
template <bool OUT_BF16, bool FUSED>
__global__ __launch_bounds__(128) void gemm_bf16_wmma(
    const unsigned short* __restrict__ A, const unsigned short* __restrict__ BT,
    int M, int N, int K, const float* __restrict__ bias,
    const float* __restrict__ scale, const float* __restrict__ sbias,
    const float* __restrict__ resid, float* __restrict__ outF,
    unsigned short* __restrict__ outB) {
  const int lane   = threadIdx.x & 31;
  const int wave   = threadIdx.x >> 5;
  const int tilesN = N >> 6;
  const int gw     = blockIdx.x * 4 + wave;
  const int mi     = gw / tilesN;
  const int ni     = gw % tilesN;
  if (mi * 16 >= M) return;               // wave-uniform; EXEC stays all-ones
  const int half = lane >> 4;
  const int nl   = lane & 15;

  const v8f zero = {0.f, 0.f, 0.f, 0.f, 0.f, 0.f, 0.f, 0.f};
  v8f acc[4] = {zero, zero, zero, zero};

  const unsigned short* arow0 = A + (size_t)(mi * 16 + nl) * K;
  for (int k0 = 0; k0 < K; k0 += 32) {
    // A fragment: lanes hold row (lane&15);
    // elements e <- K = k0 + half*16 + e (contig)
    const unsigned short* ap = arow0 + k0 + half * 16;
    __builtin_prefetch(arow0 + k0 + 64, 0, 1);   // global_prefetch next chunk
    v8bf alo = *(const v8bf*)ap;
    v8bf ahi = *(const v8bf*)(ap + 8);
    v16bf af;
#pragma unroll
    for (int i = 0; i < 8; ++i) { af[i] = alo[i]; af[i + 8] = ahi[i]; }
#pragma unroll
    for (int t = 0; t < 4; ++t) {
      // B fragment: lane column n, elements e <- K = k0 + half*16 + e (contig)
      const unsigned short* bp =
          BT + (size_t)(ni * 64 + t * 16 + nl) * K + k0 + half * 16;
      v16bf bf = *(const v16bf*)bp;
      acc[t] = __builtin_amdgcn_wmma_f32_16x16x32_bf16(
          false, af, false, bf, (short)0, acc[t], false, false);
    }
  }

#pragma unroll
  for (int t = 0; t < 4; ++t) {
#pragma unroll
    for (int r = 0; r < 8; ++r) {
      const int row = mi * 16 + half * 8 + r;   // C layout: half selects M 0-7/8-15
      const int col = ni * 64 + t * 16 + nl;
      float v = acc[t][r] + bias[col];
      if (FUSED) v = v * scale[col] + sbias[col] + resid[(size_t)row * N + col];
      if (OUT_BF16)
        outB[(size_t)row * N + col] = f2bf(v);
      else
        outF[(size_t)row * N + col] = v;
    }
  }
}

// ---------------------------------------------------------------------------
// Flash attention: 4 waves per block share one (b,h); each wave owns a
// 16-query tile. Per 32-key chunk the K-tile [32key x 64hd] and V^T-tile
// [64hd x 32key] are staged into LDS with async b128 loads (ASYNCcnt),
// then consumed as WMMA B-fragments. Online softmax uses 16-lane-half
// shuffle reductions matching the C-fragment layout; P goes through LDS
// (C-layout -> A-layout) for the PV WMMAs.
// ---------------------------------------------------------------------------
__global__ __launch_bounds__(128) void attn_kernel(
    const unsigned short* __restrict__ qkvb, const unsigned short* __restrict__ vT,
    const float* __restrict__ alibi, const unsigned char* __restrict__ mask,
    unsigned short* __restrict__ obuf) {
  __shared__ unsigned short lds_k[32 * 64];     // [key][hd]   4KB
  __shared__ unsigned short lds_v[64 * 32];     // [hd][key]   4KB
  __shared__ unsigned short lds_p[4][16 * 32];  // per-wave P  4KB
  const int tid  = threadIdx.x;
  const int lane = tid & 31;
  const int wave = tid >> 5;
  const int half = lane >> 4;
  const int nl   = lane & 15;
  const int blk  = blockIdx.x;                  // BATCH*H*(S/64) blocks
  const int q64  = blk & (S / 64 - 1);
  const int h    = (blk >> 5) & (H - 1);
  const int b    = blk >> 7;
  const int q0   = q64 * 64 + wave * 16;        // this wave's query tile
  const float scl = 0.0625f;                    // D^-0.5 (full-dim, per ref)

  // Q fragments for hd chunks [0,32) and [32,64)
  const unsigned short* qrow =
      qkvb + (size_t)(b * S + q0 + nl) * QKVW + h * 192;
  v16bf qf[2];
#pragma unroll
  for (int c = 0; c < 2; ++c) {
    v8bf lo = *(const v8bf*)(qrow + c * 32 + half * 16);
    v8bf hi = *(const v8bf*)(qrow + c * 32 + 8 + half * 16);
#pragma unroll
    for (int i = 0; i < 8; ++i) { qf[c][i] = lo[i]; qf[c][i + 8] = hi[i]; }
  }

  const v8f zero = {0.f, 0.f, 0.f, 0.f, 0.f, 0.f, 0.f, 0.f};
  v8f o[4] = {zero, zero, zero, zero};
  float m[8], l[8];
#pragma unroll
  for (int r = 0; r < 8; ++r) { m[r] = -3.0e38f; l[r] = 0.f; }

  const float* ahead = alibi + (size_t)h * S * S;
  const unsigned short* kbase = qkvb + (size_t)(b * S) * QKVW + h * 192 + 64;
  const unsigned short* vbase = vT + (size_t)(b * H + h) * HD * S;

  for (int kc = 0; kc < S; kc += 32) {
    // ---- async stage K tile (256 x 16B) and V^T tile (256 x 16B) ----
#pragma unroll
    for (int j = 0; j < 2; ++j) {
      const int i = tid * 2 + j;                     // 0..255 unique per lane
      const int krow = i >> 3, kseg = i & 7;
      async_b128((unsigned)(uintptr_t)&lds_k[krow * 64 + kseg * 8],
                 kbase + (size_t)(kc + krow) * QKVW + kseg * 8);
      const int vrow = i >> 2, vseg = i & 3;
      async_b128((unsigned)(uintptr_t)&lds_v[vrow * 32 + vseg * 8],
                 vbase + (size_t)vrow * S + kc + vseg * 8);
    }
    asm volatile("s_wait_asynccnt 0x0" ::: "memory");
    __syncthreads();

    // ---- scores: Q [16x64] x K^T -> two 16x16 tiles ----
    v8f s0 = zero, s1 = zero;
#pragma unroll
    for (int c = 0; c < 2; ++c) {
      v16bf bk0 = *(const v16bf*)&lds_k[nl * 64 + c * 32 + half * 16];
      s0 = __builtin_amdgcn_wmma_f32_16x16x32_bf16(false, qf[c], false, bk0,
                                                   (short)0, s0, false, false);
      v16bf bk1 = *(const v16bf*)&lds_k[(16 + nl) * 64 + c * 32 + half * 16];
      s1 = __builtin_amdgcn_wmma_f32_16x16x32_bf16(false, qf[c], false, bk1,
                                                   (short)0, s1, false, false);
    }

    const int key0 = kc + nl, key1 = kc + 16 + nl;
    const bool mk0 = mask[b * S + key0] != 0;
    const bool mk1 = mask[b * S + key1] != 0;
#pragma unroll
    for (int r = 0; r < 8; ++r) {
      const int qg = q0 + half * 8 + r;
      float v0 = s0[r] * scl + ahead[(size_t)qg * S + key0];
      float v1 = s1[r] * scl + ahead[(size_t)qg * S + key1];
      if (!mk0) v0 = -3.0e38f;
      if (!mk1) v1 = -3.0e38f;
      float mx = fmaxf(v0, v1);
#pragma unroll
      for (int d = 1; d < 16; d <<= 1) mx = fmaxf(mx, __shfl_xor(mx, d, 32));
      const float mn = fmaxf(m[r], mx);
      const float alpha = __expf(m[r] - mn);
      m[r] = mn;
      v0 = __expf(v0 - mn);
      v1 = __expf(v1 - mn);
      float ps = v0 + v1;
#pragma unroll
      for (int d = 1; d < 16; d <<= 1) ps += __shfl_xor(ps, d, 32);
      l[r] = l[r] * alpha + ps;
#pragma unroll
      for (int t = 0; t < 4; ++t) o[t][r] *= alpha;
      lds_p[wave][(half * 8 + r) * 32 + nl]      = f2bf(v0);
      lds_p[wave][(half * 8 + r) * 32 + 16 + nl] = f2bf(v1);
    }

    // ---- P as A fragment (16x32, K = keys), then PV WMMAs ----
    v16bf pf;
    {
      const unsigned short* pr = &lds_p[wave][nl * 32 + half * 16];
      v8bf lo = *(const v8bf*)pr;
      v8bf hi = *(const v8bf*)(pr + 8);
#pragma unroll
      for (int i = 0; i < 8; ++i) { pf[i] = lo[i]; pf[i + 8] = hi[i]; }
    }
#pragma unroll
    for (int t = 0; t < 4; ++t) {
      v16bf vf = *(const v16bf*)&lds_v[(t * 16 + nl) * 32 + half * 16];
      o[t] = __builtin_amdgcn_wmma_f32_16x16x32_bf16(false, pf, false, vf,
                                                     (short)0, o[t], false, false);
    }
    __syncthreads();   // protect lds_k/lds_v before next chunk overwrites
  }

#pragma unroll
  for (int t = 0; t < 4; ++t) {
#pragma unroll
    for (int r = 0; r < 8; ++r) {
      const int row = b * S + q0 + half * 8 + r;
      const int col = h * HD + t * 16 + nl;
      obuf[(size_t)row * D + col] = f2bf(o[t][r] / l[r]);
    }
  }
}

// ---------------------------------------------------------------------------
// Workspace layout (bytes)
// ---------------------------------------------------------------------------
static constexpr size_t OFF_WTQKV  = 0;                        // 768*256*2
static constexpr size_t OFF_WTPROJ = 393216;                   // 256*256*2
static constexpr size_t OFF_WTFFN1 = 524288;                   // 1024*256*2
static constexpr size_t OFF_WTFFN2 = 1048576;                  // 256*512*2
static constexpr size_t OFF_XLN    = 1310720;                  // TOK*D*2
static constexpr size_t OFF_QKV    = 9699328;                  // TOK*768*2
static constexpr size_t OFF_VT     = 34865152;                 // B*H*HD*S*2
static constexpr size_t OFF_OBUF   = 43253760;                 // TOK*D*2
static constexpr size_t OFF_ATTN   = 51642368;                 // TOK*D*4
static constexpr size_t OFF_HBUF   = 68419584;                 // TOK*DFF*4
static constexpr size_t OFF_GBUF   = 135528448;                // TOK*DH*2

extern "C" void kernel_launch(void* const* d_in, const int* in_sizes, int n_in,
                              void* d_out, int out_size, void* d_ws, size_t ws_size,
                              hipStream_t stream) {
  (void)in_sizes; (void)n_in; (void)out_size; (void)ws_size;
  const float* x_in       = (const float*)d_in[0];
  const unsigned char* mk = (const unsigned char*)d_in[1];
  const float* alibi      = (const float*)d_in[2];
  const float* qkv_w      = (const float*)d_in[3];
  const float* qkv_b      = (const float*)d_in[4];
  const float* proj_w     = (const float*)d_in[5];
  const float* proj_b     = (const float*)d_in[6];
  const float* ln1_g      = (const float*)d_in[7];
  const float* ln1_b      = (const float*)d_in[8];
  const float* ln2_g      = (const float*)d_in[9];
  const float* ln2_b      = (const float*)d_in[10];
  const float* ffn1_w     = (const float*)d_in[11];
  const float* ffn1_b     = (const float*)d_in[12];
  const float* ffn2_w     = (const float*)d_in[13];
  const float* ffn2_b     = (const float*)d_in[14];
  const float* attn_scale = (const float*)d_in[15];
  const float* attn_sb    = (const float*)d_in[16];
  const float* mlp_scale  = (const float*)d_in[17];
  const float* mlp_sb     = (const float*)d_in[18];
  float* outp             = (float*)d_out;

  char* ws = (char*)d_ws;
  unsigned short* wTqkv  = (unsigned short*)(ws + OFF_WTQKV);
  unsigned short* wTproj = (unsigned short*)(ws + OFF_WTPROJ);
  unsigned short* wTffn1 = (unsigned short*)(ws + OFF_WTFFN1);
  unsigned short* wTffn2 = (unsigned short*)(ws + OFF_WTFFN2);
  unsigned short* xln    = (unsigned short*)(ws + OFF_XLN);
  unsigned short* qkvb   = (unsigned short*)(ws + OFF_QKV);
  unsigned short* vT     = (unsigned short*)(ws + OFF_VT);
  unsigned short* obuf   = (unsigned short*)(ws + OFF_OBUF);
  float*          attn_o = (float*)(ws + OFF_ATTN);
  float*          hbuf   = (float*)(ws + OFF_HBUF);
  unsigned short* gbuf   = (unsigned short*)(ws + OFF_GBUF);

  // 1) weight convert/transpose -> bf16 [N,K]
  wconv_kernel<<<(D * QKVW + 255) / 256, 256, 0, stream>>>(qkv_w, wTqkv, D, QKVW);
  wconv_kernel<<<(D * D + 255) / 256, 256, 0, stream>>>(proj_w, wTproj, D, D);
  wconv_kernel<<<(D * DFF + 255) / 256, 256, 0, stream>>>(ffn1_w, wTffn1, D, DFF);
  wconv_kernel<<<(DH * D + 255) / 256, 256, 0, stream>>>(ffn2_w, wTffn2, DH, D);

  // 2) LN1 -> bf16
  ln_bf16_kernel<<<TOK / 4, 128, 0, stream>>>(x_in, ln1_g, ln1_b, xln);

  // 3) QKV GEMM (bf16 out, +bias)
  gemm_bf16_wmma<true, false><<<(TOK / 16) * (QKVW / 64) / 4, 128, 0, stream>>>(
      xln, wTqkv, TOK, QKVW, D, qkv_b, nullptr, nullptr, nullptr, nullptr, qkvb);

  // 4) V transpose per head
  vtrans_kernel<<<(BATCH * H * HD * S) / 256, 256, 0, stream>>>(qkvb, vT);

  // 5) attention (4 q-tiles per block share LDS-staged K/V)
  attn_kernel<<<BATCH * H * (S / 64), 128, 0, stream>>>(qkvb, vT, alibi, mk, obuf);

  // 6) proj GEMM fused: (*attn_scale + attn_sb) + residual(inputs) -> f32
  gemm_bf16_wmma<false, true><<<(TOK / 16) * (D / 64) / 4, 128, 0, stream>>>(
      obuf, wTproj, TOK, D, D, proj_b, attn_scale, attn_sb, x_in, attn_o, nullptr);

  // 7) LN2 -> bf16
  ln_bf16_kernel<<<TOK / 4, 128, 0, stream>>>(attn_o, ln2_g, ln2_b, xln);

  // 8) FFN1 GEMM (f32 out, +bias)
  gemm_bf16_wmma<false, false><<<(TOK / 16) * (DFF / 64) / 4, 128, 0, stream>>>(
      xln, wTffn1, TOK, DFF, D, ffn1_b, nullptr, nullptr, nullptr, hbuf, nullptr);

  // 9) GLU -> bf16
  glu_kernel<<<(TOK * DH) / 256, 256, 0, stream>>>(hbuf, gbuf);

  // 10) FFN2 GEMM fused: (*mlp_scale + mlp_sb) + residual(attn_o) -> d_out
  gemm_bf16_wmma<false, true><<<(TOK / 16) * (D / 64) / 4, 128, 0, stream>>>(
      gbuf, wTffn2, TOK, D, DH, ffn2_b, mlp_scale, mlp_sb, attn_o, outp, nullptr);
}
